// TemporalSelfAttention_70145405878474
// MI455X (gfx1250) — matrix operand order repaired and checked by reading, not compiled, not run on hardware
//
#include <hip/hip_runtime.h>

// ---------------------------------------------------------------------------
// CDNA5 (gfx1250) bf16 WMMA implementation of temporal self-attention with
// spectral-normalized output projection. wave32, V_WMMA_F32_16X16X32_BF16.
// ---------------------------------------------------------------------------

typedef __attribute__((ext_vector_type(16))) __bf16 v16bf;
typedef __attribute__((ext_vector_type(8)))  float  v8f;

union FragBF16 {
  v16bf v;
  uint4 u[2];
};

__device__ __forceinline__ unsigned short f32_to_bf16_rne(float f) {
  union { float f; unsigned int u; } x; x.f = f;
  unsigned int r = x.u + 0x7FFFu + ((x.u >> 16) & 1u);
  return (unsigned short)(r >> 16);
}

__device__ __forceinline__ float block_reduce_sum256(float s) {
  __shared__ float red[8];
  __syncthreads();
  #pragma unroll
  for (int off = 16; off > 0; off >>= 1) s += __shfl_xor(s, off);
  if ((threadIdx.x & 31) == 0) red[threadIdx.x >> 5] = s;
  __syncthreads();
  float t = 0.f;
  #pragma unroll
  for (int i = 0; i < 8; ++i) t += red[i];
  return t;
}

__device__ __forceinline__ float block_reduce_max256(float s) {
  __shared__ float red[8];
  __syncthreads();
  #pragma unroll
  for (int off = 16; off > 0; off >>= 1) s = fmaxf(s, __shfl_xor(s, off));
  if ((threadIdx.x & 31) == 0) red[threadIdx.x >> 5] = s;
  __syncthreads();
  float t = red[0];
  #pragma unroll
  for (int i = 1; i < 8; ++i) t = fmaxf(t, red[i]);
  return t;
}

// ---------------------------------------------------------------------------
// Core GEMM:  C[z,m,n] = alpha * sum_k A[z,m,k] * B[z,n,k]  (+ bias[n])
// A, B bf16 (K contiguous, "TN"), accumulation f32 in WMMA.
// Block tile 128x128x32, 256 threads = 8 waves (2x4), wave tile 64x32.
// Requires: M,N multiples of 128; K multiple of 32; lda/ldb multiples of 8.
// ---------------------------------------------------------------------------
template<bool OUT_BF16, bool HAS_BIAS>
__global__ __launch_bounds__(256)
void gemm_tn_wmma(const unsigned short* __restrict__ A, long long strideA, int lda,
                  const unsigned short* __restrict__ Bm, long long strideB, int ldb,
                  void* __restrict__ Cv, long long strideC, int ldc,
                  const float* __restrict__ bias,
                  const float* __restrict__ alpha_ptr, float alpha_const,
                  int M, int N, int K) {
  constexpr int BM = 128, BN = 128, BK = 32;
  constexpr int LDSS = 40;  // bf16 elems per LDS row (80B: conflict-free b128 reads)
  __shared__ __align__(16) unsigned short sA[BM * LDSS];
  __shared__ __align__(16) unsigned short sB[BN * LDSS];

  const int z = blockIdx.z;
  A  += (long long)z * strideA;
  Bm += (long long)z * strideB;

  const int tid   = threadIdx.x;
  const int wave  = tid >> 5;
  const int lane  = tid & 31;
  const int waveM = wave >> 2;   // 0..1 -> 64-row half
  const int waveN = wave & 3;    // 0..3 -> 32-col quarter
  const int blockM = blockIdx.y * BM;
  const int blockN = blockIdx.x * BN;
  const int mLane = lane & 15;
  const int kHalf = lane >> 4;

  float alpha = alpha_const;
  if (alpha_ptr) alpha *= *alpha_ptr;

  v8f acc[4][2];
  const v8f zero8 = {0.f, 0.f, 0.f, 0.f, 0.f, 0.f, 0.f, 0.f};
  #pragma unroll
  for (int mi = 0; mi < 4; ++mi)
    #pragma unroll
    for (int ni = 0; ni < 2; ++ni) acc[mi][ni] = zero8;

  for (int k0 = 0; k0 < K; k0 += BK) {
    // stage global -> regs (hides latency behind previous tile's compute)
    uint4 ra[2], rb[2];
    #pragma unroll
    for (int i = 0; i < 2; ++i) {
      const int c   = tid + i * 256;        // 512 b128 chunks per tile
      const int row = c >> 2;
      const int kc  = (c & 3) * 8;
      ra[i] = *(const uint4*)(A  + (long long)(blockM + row) * lda + (k0 + kc));
      rb[i] = *(const uint4*)(Bm + (long long)(blockN + row) * ldb + (k0 + kc));
    }
    __syncthreads();
    #pragma unroll
    for (int i = 0; i < 2; ++i) {
      const int c   = tid + i * 256;
      const int row = c >> 2;
      const int kc  = (c & 3) * 8;
      *(uint4*)(sA + row * LDSS + kc) = ra[i];
      *(uint4*)(sB + row * LDSS + kc) = rb[i];
    }
    __syncthreads();

    // Fragments per ISA VGPR layouts (A: M=lane%16, K chunks h*8 & 16+h*8;
    // B: N=lane%16, K chunks h*8 & 16+h*8), then 4x2 WMMAs.
    FragBF16 fa[4], fb[2];
    #pragma unroll
    for (int mi = 0; mi < 4; ++mi) {
      const unsigned short* pa = sA + (waveM * 64 + mi * 16 + mLane) * LDSS + kHalf * 8;
      fa[mi].u[0] = *(const uint4*)(pa);
      fa[mi].u[1] = *(const uint4*)(pa + 16);
    }
    #pragma unroll
    for (int ni = 0; ni < 2; ++ni) {
      const unsigned short* pb = sB + (waveN * 32 + ni * 16 + mLane) * LDSS + kHalf * 8;
      fb[ni].u[0] = *(const uint4*)(pb);
      fb[ni].u[1] = *(const uint4*)(pb + 16);
    }
    #pragma unroll
    for (int mi = 0; mi < 4; ++mi)
      #pragma unroll
      for (int ni = 0; ni < 2; ++ni)
        acc[mi][ni] = __builtin_amdgcn_wmma_f32_16x16x32_bf16(
            false, fa[mi].v, false, fb[ni].v, (short)0, acc[mi][ni], false, false);
  }

  // Epilogue. C/D layout: lane l, vgpr r -> M = 8*(l/16)+r, N = l%16.
  const int mBase = blockM + waveM * 64;
  const int nBase = blockN + waveN * 32;
  #pragma unroll
  for (int mi = 0; mi < 4; ++mi) {
    #pragma unroll
    for (int ni = 0; ni < 2; ++ni) {
      const int n = nBase + ni * 16 + mLane;
      const float bv = HAS_BIAS ? bias[n] : 0.0f;
      #pragma unroll
      for (int r = 0; r < 8; ++r) {
        const int m = mBase + mi * 16 + kHalf * 8 + r;
        const float val = acc[mi][ni][r] * alpha + bv;
        const long long idx = (long long)z * strideC + (long long)m * ldc + n;
        if constexpr (OUT_BF16) ((unsigned short*)Cv)[idx] = f32_to_bf16_rne(val);
        else                    ((float*)Cv)[idx] = val;
      }
    }
  }
}

// ---------------------------------------------------------------------------
// Prep / pointwise kernels
// ---------------------------------------------------------------------------
__global__ void cvt_f32_bf16(const float* __restrict__ src,
                             unsigned short* __restrict__ dst, long long n) {
  long long i = (long long)blockIdx.x * 256 + threadIdx.x;
  if (i < n) dst[i] = f32_to_bf16_rne(src[i]);
}

// partial sums over T slices for the temporal mean: grid (B*C/256, slices)
__global__ void mean_partial(const float* __restrict__ x, float* __restrict__ mean,
                             int T, int C, int tlen) {
  const int idx = blockIdx.x * 256 + threadIdx.x;  // over B*C
  const int b = idx / C;
  const int c = idx - b * C;
  const int t0 = blockIdx.y * tlen;
  const float* p = x + ((long long)b * T + t0) * C + c;
  float s = 0.f;
  for (int t = 0; t < tlen; ++t) s += p[(long long)t * C];
  atomicAdd(&mean[idx], s);
}

__global__ void center_to_bf16(const float* __restrict__ x, const float* __restrict__ mean,
                               unsigned short* __restrict__ xc, int T, int C, float invT) {
  const long long i = (long long)blockIdx.x * 256 + threadIdx.x;  // over B*T*C
  const int c = (int)(i % C);
  const long long bt = i / C;
  const int b = (int)(bt / T);
  xc[i] = f32_to_bf16_rne(x[i] - mean[b * C + c] * invT);
}

// vT[b][c][t] = qkv[b*T+t][vofs+c]   (tiled LDS transpose)
__global__ void transpose_v(const unsigned short* __restrict__ qkv,
                            unsigned short* __restrict__ vT,
                            int T, int C, int ldq, int vofs) {
  __shared__ unsigned short tile[32][33];
  const int b = blockIdx.z;
  const int c0 = blockIdx.x * 32;
  const int t0 = blockIdx.y * 32;
  const unsigned short* src = qkv + (long long)b * T * ldq;
  for (int i = threadIdx.y; i < 32; i += 8)
    tile[i][threadIdx.x] = src[(long long)(t0 + i) * ldq + vofs + c0 + threadIdx.x];
  __syncthreads();
  unsigned short* dst = vT + (long long)b * C * T;
  for (int i = threadIdx.y; i < 32; i += 8)
    dst[(long long)(c0 + i) * T + t0 + threadIdx.x] = tile[threadIdx.x][i];
}

// Row softmax over T=2048 f32 scores; writes P as bf16 IN PLACE over the row.
__global__ __launch_bounds__(256)
void softmax_rows(float* __restrict__ S, int T) {
  float* srow = S + (long long)blockIdx.x * T;
  unsigned short* prow = (unsigned short*)srow;
  const int tid = threadIdx.x;
  float vals[8];
  float mx = -3.4e38f;
  #pragma unroll
  for (int i = 0; i < 8; ++i) { vals[i] = srow[tid + i * 256]; mx = fmaxf(mx, vals[i]); }
  mx = block_reduce_max256(mx);   // barrier => all reads done before any write below
  float s = 0.f;
  #pragma unroll
  for (int i = 0; i < 8; ++i) { vals[i] = __expf(vals[i] - mx); s += vals[i]; }
  s = block_reduce_sum256(s);
  const float inv = 1.0f / s;
  #pragma unroll
  for (int i = 0; i < 8; ++i) prow[tid + i * 256] = f32_to_bf16_rne(vals[i] * inv);
}

// ---------------------------------------------------------------------------
// Spectral norm: sigma = || W * l2norm(W^T u) || ; emit 1/sigma
// ---------------------------------------------------------------------------
__global__ void spect_wtu(const float* __restrict__ W, const float* __restrict__ u,
                          float* __restrict__ y, int C) {
  const int j = blockIdx.x * 256 + threadIdx.x;
  float s = 0.f;
  for (int i = 0; i < C; ++i) s += W[(long long)i * C + j] * u[i];
  y[j] = s;
}

__global__ void spect_normalize(const float* __restrict__ y, float* __restrict__ v, int C) {
  float s = 0.f;
  for (int i = threadIdx.x; i < C; i += 256) { const float t = y[i]; s += t * t; }
  const float tot = block_reduce_sum256(s);
  const float inv = 1.0f / fmaxf(sqrtf(tot), 1e-12f);
  for (int i = threadIdx.x; i < C; i += 256) v[i] = y[i] * inv;
}

__global__ void spect_wv(const float* __restrict__ W, const float* __restrict__ v,
                         float* __restrict__ z, int C) {
  const float* row = W + (long long)blockIdx.x * C;
  float s = 0.f;
  for (int j = threadIdx.x; j < C; j += 256) s += row[j] * v[j];
  s = block_reduce_sum256(s);
  if (threadIdx.x == 0) z[blockIdx.x] = s;
}

__global__ void spect_sigma(const float* __restrict__ z, float* __restrict__ inv_sigma, int C) {
  float s = 0.f;
  for (int i = threadIdx.x; i < C; i += 256) { const float t = z[i]; s += t * t; }
  const float tot = block_reduce_sum256(s);
  if (threadIdx.x == 0) {
    const float nz = sqrtf(tot);
    const float sigma = tot / fmaxf(nz, 1e-12f);  // u1 . (W v) = ||z||
    inv_sigma[0] = 1.0f / sigma;
  }
}

// ---------------------------------------------------------------------------
extern "C" void kernel_launch(void* const* d_in, const int* in_sizes, int n_in,
                              void* d_out, int out_size, void* d_ws, size_t ws_size,
                              hipStream_t stream) {
  (void)in_sizes; (void)n_in; (void)out_size; (void)ws_size;
  const int B = 8, T = 2048, C = 1024;

  const float* x      = (const float*)d_in[0];
  const float* qkv_w  = (const float*)d_in[1];
  const float* qkv_b  = (const float*)d_in[2];
  const float* out_w  = (const float*)d_in[3];
  const float* out_b  = (const float*)d_in[4];
  const float* u_vec  = (const float*)d_in[5];

  char* ws = (char*)d_ws;
  size_t off = 0;
  auto take = [&](size_t bytes) -> char* {
    char* p = ws + off;
    off = (off + bytes + 255) & ~(size_t)255;
    return p;
  };
  unsigned short* xc     = (unsigned short*)take((size_t)B * T * C * 2);       // 32 MB
  unsigned short* qkv    = (unsigned short*)take((size_t)B * T * 3 * C * 2);   // 96 MB
  unsigned short* vT     = (unsigned short*)take((size_t)B * C * T * 2);       // 32 MB
  float*          S      = (float*)take((size_t)B * T * T * 4);                // 128 MB
  unsigned short* aout   = (unsigned short*)take((size_t)B * T * C * 2);       // 32 MB
  unsigned short* wqkvbf = (unsigned short*)take((size_t)3 * C * C * 2);       // 6 MB
  unsigned short* woutbf = (unsigned short*)take((size_t)C * C * 2);           // 2 MB
  float*          meanb  = (float*)take((size_t)B * C * 4);
  float*          yv     = (float*)take((size_t)C * 4);
  float*          vv     = (float*)take((size_t)C * 4);
  float*          zv     = (float*)take((size_t)C * 4);
  float*          invsig = (float*)take(256);

  const float inv_sqrt_c = 1.0f / sqrtf((float)C);

  // weights -> bf16
  cvt_f32_bf16<<<(3 * C * C) / 256, 256, 0, stream>>>(qkv_w, wqkvbf, (long long)3 * C * C);
  cvt_f32_bf16<<<(C * C) / 256, 256, 0, stream>>>(out_w, woutbf, (long long)C * C);

  // temporal mean-center -> bf16
  hipMemsetAsync(meanb, 0, (size_t)B * C * 4, stream);
  mean_partial<<<dim3(B * C / 256, 16), 256, 0, stream>>>(x, meanb, T, C, T / 16);
  center_to_bf16<<<(int)(((long long)B * T * C) / 256), 256, 0, stream>>>(
      x, meanb, xc, T, C, 1.0f / (float)T);

  // QKV projection: [B*T, 3C] = xc @ qkv_w^T + qkv_b
  gemm_tn_wmma<true, true><<<dim3(3 * C / 128, (B * T) / 128, 1), 256, 0, stream>>>(
      xc, 0, C, wqkvbf, 0, C, qkv, 0, 3 * C, qkv_b, nullptr, 1.0f, B * T, 3 * C, C);

  // V -> vT [B, C, T] so P@V is also a TN GEMM
  transpose_v<<<dim3(C / 32, T / 32, B), dim3(32, 8), 0, stream>>>(qkv, vT, T, C, 3 * C, 2 * C);

  // scores: S[b] = (Q K^T) * C^-0.5   (Q,K strided views into qkv)
  gemm_tn_wmma<false, false><<<dim3(T / 128, T / 128, B), 256, 0, stream>>>(
      qkv + 0, (long long)T * 3 * C, 3 * C,
      qkv + C, (long long)T * 3 * C, 3 * C,
      S, (long long)T * T, T, nullptr, nullptr, inv_sqrt_c, T, T, C);

  // softmax rows (P written in place as bf16, row stride 2T bf16)
  softmax_rows<<<B * T, 256, 0, stream>>>(S, T);

  // attention output: aout[b] = P[b] @ V[b]  == TN with B = vT
  gemm_tn_wmma<true, false><<<dim3(C / 128, T / 128, B), 256, 0, stream>>>(
      (unsigned short*)S, (long long)2 * T * T, 2 * T,
      vT, (long long)C * T, T,
      aout, (long long)T * C, C, nullptr, nullptr, 1.0f, T, C, T);

  // spectral norm -> 1/sigma
  spect_wtu<<<C / 256, 256, 0, stream>>>(out_w, u_vec, yv, C);
  spect_normalize<<<1, 256, 0, stream>>>(yv, vv, C);
  spect_wv<<<C, 256, 0, stream>>>(out_w, vv, zv, C);
  spect_sigma<<<1, 256, 0, stream>>>(zv, invsig, C);

  // output projection: d_out = (aout @ out_w^T) * (1/sigma) + out_b  (f32)
  gemm_tn_wmma<false, true><<<dim3(C / 128, (B * T) / 128, 1), 256, 0, stream>>>(
      aout, 0, C, woutbf, 0, C, (float*)d_out, 0, C, out_b, invsig, 1.0f, B * T, C, C);
}
